// Pointer_9620726743383
// MI455X (gfx1250) — compile-verified, matrix operand order checked by reading, not run on hardware
//
#include <hip/hip_runtime.h>
#include <hip/hip_bf16.h>
#include <math.h>

#define BD 128
#define HD 512
#define SD 2048

typedef __attribute__((ext_vector_type(16))) __bf16 v16bf;
typedef __attribute__((ext_vector_type(8)))  float  v8f;

union Frag16 { v16bf v; uint4 q[2]; };

__device__ __forceinline__ unsigned short f2bf(float f) {
    union { float f; unsigned u; } v; v.f = f;
    unsigned r = v.u + 0x7FFFu + ((v.u >> 16) & 1u);   // round-to-nearest-even
    return (unsigned short)(r >> 16);
}

// ---------------------------------------------------------------- W -> bf16
__global__ __launch_bounds__(256)
void kConvW(const float* __restrict__ w, unsigned short* __restrict__ wbf) {
    int i = blockIdx.x * 1024 + threadIdx.x;
#pragma unroll
    for (int j = 0; j < 4; ++j) wbf[i + j * 256] = f2bf(w[i + j * 256]);
}

// ------------------------------------------------- t[b,s] = v_att . tanh(W @ hid)
// block: 512 threads (16 waves). Covers all 32 h-tiles (2 per wave), 64 s-cols.
__global__ __launch_bounds__(512)
void kScores(const float* __restrict__ hid, const unsigned short* __restrict__ wbf,
             const float* __restrict__ v_att, float* __restrict__ tout)
{
    __shared__ __align__(16) unsigned short sh[64 * 264];  // hidden tile, [s][k] bf16
    __shared__ float shT[64];

    const int tid  = threadIdx.x;
    const int b    = blockIdx.y;
    const int sb   = blockIdx.x * 64;
    const int wave = tid >> 5, lane = tid & 31;
    const int half = lane >> 4, lme = lane & 15;
    const int ht0  = wave * 2;                // this wave's first h-tile

    if (tid < 64) shT[tid] = 0.f;

    v8f acc[2][4] = {};                       // 2 h-tiles x 4 s-tiles, f32 C/D

    const float* hb = hid + (size_t)b * HD * SD;

    for (int kh = 0; kh < 2; ++kh) {          // two 256-wide k halves
        const int k0 = kh * 256;
        __syncthreads();                      // WAR on sh (and shT zero, 1st round)
        // stage hidden[b, k0..k0+255, sb..sb+63] -> sh[s*264 + k] (bf16)
#pragma unroll 4
        for (int i = 0; i < 32; ++i) {
            int idx = i * 512 + tid;
            int s   = idx & 63;               // lanes consecutive in s -> coalesced
            int kk  = idx >> 6;
            sh[s * 264 + kk] = f2bf(hb[(size_t)(k0 + kk) * SD + sb + s]);
        }
        __syncthreads();

        for (int kb = 0; kb < 8; ++kb) {      // k-blocks of 32 within this half
            const int kkb = kb * 32 + 8 * half;
            Frag16 Bf[4];
#pragma unroll
            for (int st = 0; st < 4; ++st) {  // B: per-lane column of hidden
                const int n = st * 16 + lme;
                Bf[st].q[0] = *(const uint4*)&sh[n * 264 + kkb];
                Bf[st].q[1] = *(const uint4*)&sh[n * 264 + kkb + 16];
            }
#pragma unroll
            for (int t = 0; t < 2; ++t) {
                const int h  = (ht0 + t) * 16 + lme;   // A: per-lane row of W
                const int kg = k0 + kkb;
                Frag16 Af;
                Af.q[0] = *(const uint4*)&wbf[(size_t)h * HD + kg];
                Af.q[1] = *(const uint4*)&wbf[(size_t)h * HD + kg + 16];
#pragma unroll
                for (int st = 0; st < 4; ++st) {
                    acc[t][st] = __builtin_amdgcn_wmma_f32_16x16x32_bf16(
                        false, Af.v, false, Bf[st].v,
                        (short)0, acc[t][st], false, false);
                }
            }
        }
    }
    __syncthreads();

    // epilogue: part[st] = sum over this wave's 32 h of v_att[h]*tanh(score)
    float part[4] = {0.f, 0.f, 0.f, 0.f};
#pragma unroll
    for (int t = 0; t < 2; ++t) {
#pragma unroll
        for (int r = 0; r < 8; ++r) {
            const int   h  = (ht0 + t) * 16 + 8 * half + r;  // C layout: half->M+8
            const float va = v_att[h];
#pragma unroll
            for (int st = 0; st < 4; ++st)
                part[st] += va * tanhf(acc[t][st][r]);
        }
    }
#pragma unroll
    for (int st = 0; st < 4; ++st) part[st] += __shfl_xor(part[st], 16, 32);
    if (lane < 16) {
#pragma unroll
        for (int st = 0; st < 4; ++st) atomicAdd(&shT[st * 16 + lme], part[st]);
    }
    __syncthreads();
    if (tid < 64) tout[(size_t)b * SD + sb + tid] = shT[tid];
}

// ---------------------------------------------------------------- softmax rows
__global__ __launch_bounds__(256)
void kSoftmax(float* __restrict__ t) {
    __shared__ float red[256];
    const int b = blockIdx.x, tid = threadIdx.x;
    float* row = t + (size_t)b * SD;
    float m = -1e30f;
    for (int i = tid; i < SD; i += 256) m = fmaxf(m, row[i]);
    red[tid] = m; __syncthreads();
    for (int o = 128; o > 0; o >>= 1) {
        if (tid < o) red[tid] = fmaxf(red[tid], red[tid + o]);
        __syncthreads();
    }
    m = red[0]; __syncthreads();
    float s = 0.f;
    for (int i = tid; i < SD; i += 256) { float e = __expf(row[i] - m); row[i] = e; s += e; }
    red[tid] = s; __syncthreads();
    for (int o = 128; o > 0; o >>= 1) {
        if (tid < o) red[tid] += red[tid + o];
        __syncthreads();
    }
    const float inv = 1.f / red[0];
    for (int i = tid; i < SD; i += 256) row[i] *= inv;
}

// ------------------------------------------- context[b,h] = attn . hidden[b,h,:]
__global__ __launch_bounds__(256)
void kContext(const float* __restrict__ hid, const float* __restrict__ attn,
              float* __restrict__ ctx) {
    const int b = blockIdx.y;
    const int h = blockIdx.x * 8 + (threadIdx.x >> 5);
    const int lane = threadIdx.x & 31;
    const float* hp = hid + ((size_t)b * HD + h) * SD;
    const float* ap = attn + (size_t)b * SD;
    float acc = 0.f;
    for (int i = lane; i < SD; i += 32) acc += ap[i] * hp[i];
#pragma unroll
    for (int o = 16; o > 0; o >>= 1) acc += __shfl_xor(acc, o, 32);
    if (lane == 0) ctx[(size_t)b * HD + h] = acc;
}

// ---------------------------------------------------------------- 2-layer MLP
__global__ __launch_bounds__(256)
void kMLP(const float* __restrict__ ctx,
          const float* __restrict__ w1, const float* __restrict__ b1,
          const float* __restrict__ w2, const float* __restrict__ b2,
          float* __restrict__ outb) {
    __shared__ float cx[HD], h1[HD];
    const int b = blockIdx.x, tid = threadIdx.x;
    const int wave = tid >> 5, lane = tid & 31;
    cx[tid]       = ctx[(size_t)b * HD + tid];
    cx[tid + 256] = ctx[(size_t)b * HD + tid + 256];
    __syncthreads();
    for (int jj = 0; jj < 64; ++jj) {           // one output per wave per iter
        const int j = wave * 64 + jj;
        const float* w = w1 + (size_t)j * HD;
        float a = 0.f;
        for (int k = lane; k < HD; k += 32) a += cx[k] * w[k];
#pragma unroll
        for (int o = 16; o > 0; o >>= 1) a += __shfl_xor(a, o, 32);
        if (lane == 0) h1[j] = fmaxf(a + b1[j], 0.f);
    }
    __syncthreads();
    for (int jj = 0; jj < 64; ++jj) {
        const int j = wave * 64 + jj;
        const float* w = w2 + (size_t)j * HD;
        float a = 0.f;
        for (int k = lane; k < HD; k += 32) a += h1[k] * w[k];
#pragma unroll
        for (int o = 16; o > 0; o >>= 1) a += __shfl_xor(a, o, 32);
        if (lane == 0) outb[(size_t)b * HD + j] = fmaxf(a + b2[j], 0.f);
    }
}

// ---------------------- probs[b,s] = v_ptr . tanh(hidden[b,:,s] + out[b,:])
__global__ __launch_bounds__(256)
void kProbs(const float* __restrict__ hid, const float* __restrict__ outb,
            const float* __restrict__ v_ptr, float* __restrict__ probs) {
    const int b = blockIdx.y;
    const int s = blockIdx.x * 256 + threadIdx.x;
    const float* hp = hid + (size_t)b * HD * SD + s;
    const float* ob = outb + (size_t)b * HD;
    float acc = 0.f;
#pragma unroll 4
    for (int h = 0; h < HD; ++h) {
        __builtin_prefetch(&hp[(size_t)(h + 8) * SD], 0, 1);
        acc += v_ptr[h] * tanhf(hp[(size_t)h * SD] + ob[h]);
    }
    probs[(size_t)b * SD + s] = acc;
}

extern "C" void kernel_launch(void* const* d_in, const int* in_sizes, int n_in,
                              void* d_out, int out_size, void* d_ws, size_t ws_size,
                              hipStream_t stream) {
    const float* hid   = (const float*)d_in[0];
    const float* v_att = (const float*)d_in[1];
    const float* W_att = (const float*)d_in[2];
    const float* v_ptr = (const float*)d_in[3];
    const float* fc1w  = (const float*)d_in[4];
    const float* fc1b  = (const float*)d_in[5];
    const float* fc2w  = (const float*)d_in[6];
    const float* fc2b  = (const float*)d_in[7];
    float* probs = (float*)d_out;

    char* ws = (char*)d_ws;
    unsigned short* wbf = (unsigned short*)ws;                         // 512 KB
    float* t    = (float*)(ws + (512 << 10));                          // 1 MB (t -> attn)
    float* ctx  = (float*)(ws + (512 << 10) + (1 << 20));              // 256 KB
    float* outb = (float*)(ws + (512 << 10) + (1 << 20) + (256 << 10));// 256 KB

    kConvW  <<<256, 256, 0, stream>>>(W_att, wbf);
    kScores <<<dim3(SD / 64, BD), 512, 0, stream>>>(hid, wbf, v_att, t);
    kSoftmax<<<BD, 256, 0, stream>>>(t);
    kContext<<<dim3(HD / 8, BD), 256, 0, stream>>>(hid, t, ctx);
    kMLP    <<<BD, 256, 0, stream>>>(ctx, fc1w, fc1b, fc2w, fc2b, outb);
    kProbs  <<<dim3(SD / 256, BD), 256, 0, stream>>>(hid, outb, v_ptr, probs);
}
